// SRCNN_27556510171587
// MI455X (gfx1250) — matrix operand order repaired and checked by reading, not compiled, not run on hardware
//
#include <hip/hip_runtime.h>

typedef __attribute__((ext_vector_type(16))) __bf16 bf16x16;
typedef __attribute__((ext_vector_type(8)))  float  f32x8;

#define H_LAT 181
#define W_LON 360
#define BATCH 8
#define NSPAT (BATCH * H_LAT * W_LON)   // 521280 spatial points

// ---------- helpers ----------
static __device__ inline unsigned short f2bf(float f) {
  __bf16 h = (__bf16)f;
  unsigned short s;
  __builtin_memcpy(&s, &h, 2);
  return s;
}
static __device__ inline float bflo(unsigned u) { return __uint_as_float(u << 16); }
static __device__ inline float bfhi(unsigned u) { return __uint_as_float(u & 0xffff0000u); }

// Low 32 bits of a generic shared-memory address == LDS byte offset
// (flat aperture: LDS address is the low 32 bits of the flat address, ISA 10.2).
static __device__ inline unsigned lds_off(const void* p) {
  return (unsigned)(unsigned long long)p;
}
// CDNA5 async copy: global memory -> LDS, 16 bytes per lane, ASYNCcnt-tracked.
static __device__ inline void async_copy_b128(unsigned lds_addr, const void* gaddr) {
  asm volatile("global_load_async_to_lds_b128 %0, %1, off"
               :: "v"(lds_addr), "v"(gaddr)
               : "memory");
}
static __device__ inline void async_wait0() {
  asm volatile("s_wait_asynccnt 0x0" ::: "memory");
}

// A fragment (16x32 bf16) from row-major [M][K] storage, K contiguous.
// lanes<16 hold K=0..15, lanes>=16 hold K=16..31 (matches frag_b).
static __device__ inline bf16x16 frag_a(const unsigned* row, int hi) {
  union { bf16x16 v; unsigned u[8]; } r;
  const int kb = hi * 8;
  r.u[0] = row[kb + 0]; r.u[1] = row[kb + 1];
  r.u[2] = row[kb + 2]; r.u[3] = row[kb + 3];
  r.u[4] = row[kb + 4]; r.u[5] = row[kb + 5];
  r.u[6] = row[kb + 6]; r.u[7] = row[kb + 7];
  return r.v;
}
// B fragment (32x16 bf16) from row-major [N][K] storage (K contiguous):
// lanes<16 hold K=0..15 of column n=lane, lanes>=16 hold K=16..31 of n=lane-16.
static __device__ inline bf16x16 frag_b(const unsigned* row, int hi) {
  union { bf16x16 v; unsigned u[8]; } r;
  const int kb = hi * 8;
#pragma unroll
  for (int i = 0; i < 8; ++i) r.u[i] = row[kb + i];
  return r.v;
}
static __device__ inline f32x8 wmma_bf16(bf16x16 a, bf16x16 b, f32x8 c) {
  return __builtin_amdgcn_wmma_f32_16x16x32_bf16(false, a, false, b, (short)0, c,
                                                 false, false);
}

// ---------- weight prep ----------
// w1 [128][3][9] f32 -> w1p [128][32] bf16 (K = c*9+k, padded 27..31 = 0)
__global__ void k_prep_w1(const float* __restrict__ w1, __bf16* __restrict__ w1p) {
  const int o = threadIdx.x;  // 128 threads
  for (int ck = 0; ck < 32; ++ck) {
    float v = (ck < 27) ? w1[(o * 3 + ck / 9) * 9 + (ck % 9)] : 0.f;
    w1p[o * 32 + ck] = (__bf16)v;
  }
}
// w2 [64][128][5][5] f32 -> w2t [64][25][128] bf16 (channel-contiguous per tap)
__global__ __launch_bounds__(256) void k_prep_w2(const float* __restrict__ w2,
                                                 __bf16* __restrict__ w2t) {
  const int idx = blockIdx.x * 256 + threadIdx.x;   // 204800 total, exact
  const int o = idx / 3200, tap = (idx / 128) % 25, c = idx % 128;
  w2t[idx] = (__bf16)w2[(o * 128 + c) * 25 + tap];
}

// ---------- layer 1 stage A: gather + psi contraction ----------
// z[n][ck] bf16, n = ((b*181+h)*360+w), ck = c*9+k (27 real, 5 zero pad)
__global__ __launch_bounds__(256) void k_disco_z(const float* __restrict__ x,
                                                 const float* __restrict__ psi,
                                                 const float* __restrict__ quad,
                                                 __bf16* __restrict__ z) {
  const int tid = blockIdx.x * 256 + threadIdx.x;
  const int ck = tid & 31;
  const int n = tid >> 5;
  const int w = n % W_LON;
  const int h = (n / W_LON) % H_LAT;
  const int b = n / (W_LON * H_LAT);
  float acc = 0.f;
  if (ck < 27) {
    const int c = ck / 9;
    const int kk = ck % 9;
    const float* xb = x + (size_t)(b * 3 + c) * (H_LAT * W_LON);
    const float* pp = psi + ((size_t)kk * H_LAT + h) * 25;
#pragma unroll
    for (int i = 0; i < 5; ++i) {
      int hi = h + i - 2;
      hi = hi < 0 ? 0 : (hi > H_LAT - 1 ? H_LAT - 1 : hi);
      const float qv = quad[hi];
      const float* xr = xb + (size_t)hi * W_LON;
      float s = 0.f;
#pragma unroll
      for (int j = 0; j < 5; ++j) {
        int wj = w + j - 2;
        wj += (wj < 0) ? W_LON : 0;
        wj -= (wj >= W_LON) ? W_LON : 0;
        s += pp[i * 5 + j] * xr[wj];
      }
      acc += qv * s;
    }
  }
  ((unsigned short*)z)[(size_t)n * 32 + ck] = f2bf(acc);
}

// ---------- layer 1 stage B: 128x27 GEMM via WMMA (single k-step) ----------
// h1[n][128] bf16 = relu(w1p @ z + b1), Ntile = 64, 8 waves (one M-subtile each)
__global__ __launch_bounds__(256) void k_gemm1(const __bf16* __restrict__ z,
                                               const __bf16* __restrict__ w1p,
                                               const float* __restrict__ b1,
                                               __bf16* __restrict__ h1) {
  __shared__ __bf16 zt[64 * 32];   // 4 KB tile of B
  const int tid = threadIdx.x;
  const int n0 = blockIdx.x * 64;
  // async stage: one 16B group per thread, straight into LDS
  async_copy_b128(lds_off(zt) + (unsigned)(tid * 16),
                  (const unsigned char*)z + (size_t)n0 * 64 + (size_t)tid * 16);
  async_wait0();
  __syncthreads();

  const int wid = tid >> 5, lane = tid & 31;
  const int m = lane & 15, hi = lane >> 4;
  const int mi = wid;  // 0..7 -> 128 output channels
  const unsigned* arow = (const unsigned*)(w1p + (mi * 16 + m) * 32);
  const bf16x16 a = frag_a(arow, hi);

  f32x8 acc[4];
#pragma unroll
  for (int ns = 0; ns < 4; ++ns) {
    const unsigned* brow = (const unsigned*)(zt + (ns * 16 + m) * 32);
    f32x8 c = {};
    acc[ns] = wmma_bf16(a, frag_b(brow, hi), c);
  }
  const int obase = mi * 16 + hi * 8;
#pragma unroll
  for (int ns = 0; ns < 4; ++ns) {
    const size_t n = (size_t)(n0 + ns * 16 + m);
    unsigned q[4];
#pragma unroll
    for (int r = 0; r < 4; ++r) {
      float v0 = acc[ns][2 * r] + b1[obase + 2 * r];
      float v1 = acc[ns][2 * r + 1] + b1[obase + 2 * r + 1];
      v0 = v0 > 0.f ? v0 : 0.f;
      v1 = v1 > 0.f ? v1 : 0.f;
      q[r] = (unsigned)f2bf(v0) | ((unsigned)f2bf(v1) << 16);
    }
    *(uint4*)((unsigned short*)h1 + n * 128 + obase) = make_uint4(q[0], q[1], q[2], q[3]);
  }
}

// ---------- layer 2: 5x5 conv 128->64 as 25 shifted WMMA GEMMs (dominant) ----------
// grid (3, 181, 8); 8 waves; Mtile=64, Ntile=128
// LDS: slab [5][132][128] bf16 + double-buffered per-tap weights 2x[64][128] bf16
__global__ __launch_bounds__(256) void k_conv2(const __bf16* __restrict__ h1,
                                               const __bf16* __restrict__ w2t,
                                               const float* __restrict__ b2,
                                               __bf16* __restrict__ h2) {
  extern __shared__ __bf16 smem[];
  __bf16* slab = smem;                    // [5][132][128]
  __bf16* wlbase = smem + 5 * 132 * 128;  // 2 x [64][128], selected by (tap & 1)
  const int tid = threadIdx.x;
  const int w0 = blockIdx.x * 128;
  const int h = blockIdx.y;
  const int b = blockIdx.z;

  // prefetch tap-0 weights (async, global -> LDS)
  {
    const unsigned dst = lds_off(wlbase);
#pragma unroll
    for (int g = tid; g < 1024; g += 256)
      async_copy_b128(dst + (unsigned)(g * 16),
                      (const unsigned char*)w2t + (size_t)(g >> 4) * 25 * 256 +
                          (size_t)(g & 15) * 16);
  }
  // stage input slab: async for in-bounds groups, ds zero-store for padding
  {
    const unsigned dst = lds_off(slab);
    for (int g = tid; g < 5 * 132 * 16; g += 256) {
      const int cg = g & 15;
      const int col = (g >> 4) % 132;
      const int r = (g >> 4) / 132;
      const int hr = h + r - 2;
      const int wc = w0 + col - 2;
      const unsigned laddr = dst + (unsigned)(((r * 132 + col) * 128 + cg * 8) * 2);
      if (hr >= 0 && hr < H_LAT && wc >= 0 && wc < W_LON) {
        async_copy_b128(laddr,
                        (const unsigned short*)h1 +
                            (((size_t)b * H_LAT + hr) * W_LON + wc) * 128 + cg * 8);
      } else {
        *(uint4*)((unsigned short*)slab + ((size_t)(r * 132 + col)) * 128 + cg * 8) =
            make_uint4(0u, 0u, 0u, 0u);
      }
    }
  }
  async_wait0();
  __syncthreads();

  const int wid = tid >> 5, lane = tid & 31;
  const int m = lane & 15, hif = lane >> 4;
  const int mi = wid & 3, nh = wid >> 2;
  f32x8 acc[4] = {};

  for (int tap = 0; tap < 25; ++tap) {
    const int ti = tap / 5, tj = tap % 5;
    const __bf16* wlc = wlbase + (tap & 1) * (64 * 128);
    // prefetch next tap's weights into the other buffer while we compute
    if (tap < 24) {
      const unsigned dst = lds_off(wlbase + ((tap + 1) & 1) * (64 * 128));
#pragma unroll
      for (int g = tid; g < 1024; g += 256)
        async_copy_b128(dst + (unsigned)(g * 16),
                        (const unsigned char*)w2t + (size_t)(g >> 4) * 25 * 256 +
                            (size_t)(tap + 1) * 256 + (size_t)(g & 15) * 16);
    }
#pragma unroll
    for (int kc = 0; kc < 4; ++kc) {
      const unsigned* arow = (const unsigned*)(wlc + (mi * 16 + m) * 128 + kc * 32);
      const bf16x16 a = frag_a(arow, hif);
#pragma unroll
      for (int ns = 0; ns < 4; ++ns) {
        const int scol = nh * 64 + ns * 16 + m + tj;   // <= 131
        const unsigned* brow =
            (const unsigned*)(slab + ((size_t)(ti * 132 + scol)) * 128 + kc * 32);
        acc[ns] = wmma_bf16(a, frag_b(brow, hif), acc[ns]);
      }
    }
    async_wait0();       // next-tap weights landed (cheap: overlapped with WMMAs)
    __syncthreads();     // all waves done reading wlc before it is overwritten
  }

  const int obase = mi * 16 + hif * 8;
#pragma unroll
  for (int ns = 0; ns < 4; ++ns) {
    const int wout = w0 + nh * 64 + ns * 16 + m;
    if (wout < W_LON) {
      unsigned q[4];
#pragma unroll
      for (int r = 0; r < 4; ++r) {
        float v0 = acc[ns][2 * r] + b2[obase + 2 * r];
        float v1 = acc[ns][2 * r + 1] + b2[obase + 2 * r + 1];
        v0 = v0 > 0.f ? v0 : 0.f;
        v1 = v1 > 0.f ? v1 : 0.f;
        q[r] = (unsigned)f2bf(v0) | ((unsigned)f2bf(v1) << 16);
      }
      *(uint4*)((unsigned short*)h2 +
                (((size_t)b * H_LAT + h) * W_LON + wout) * 64 + obase) =
          make_uint4(q[0], q[1], q[2], q[3]);
    }
  }
}

// ---------- layer 3: 5x5 conv 64->3, f32 VALU (M=3 too small for WMMA) ----------
__global__ __launch_bounds__(256) void k_conv3(const __bf16* __restrict__ h2,
                                               const float* __restrict__ w3,
                                               const float* __restrict__ b3,
                                               float* __restrict__ out) {
  __shared__ float lw[3 * 64 * 25];   // 18.75 KB, same linear layout as w3
  for (int i = threadIdx.x; i < 3 * 64 * 25; i += 256) lw[i] = w3[i];
  __syncthreads();
  const int n = blockIdx.x * 256 + threadIdx.x;
  if (n >= NSPAT) return;
  const int w = n % W_LON;
  const int h = (n / W_LON) % H_LAT;
  const int b = n / (W_LON * H_LAT);
  float a0 = 0.f, a1 = 0.f, a2 = 0.f;
  for (int ti = 0; ti < 5; ++ti) {
    const int hr = h + ti - 2;
    if (hr < 0 || hr >= H_LAT) continue;
#pragma unroll
    for (int tj = 0; tj < 5; ++tj) {
      const int wc = w + tj - 2;
      if (wc < 0 || wc >= W_LON) continue;
      const int tap = ti * 5 + tj;
      const uint4* row = (const uint4*)((const unsigned short*)h2 +
                                        (((size_t)b * H_LAT + hr) * W_LON + wc) * 64);
#pragma unroll
      for (int cg = 0; cg < 8; ++cg) {
        const uint4 qv = row[cg];
        const unsigned uu[4] = {qv.x, qv.y, qv.z, qv.w};
#pragma unroll
        for (int e = 0; e < 4; ++e) {
          const int c = cg * 8 + e * 2;
          const float f0 = bflo(uu[e]);
          const float f1 = bfhi(uu[e]);
          a0 += f0 * lw[(0 * 64 + c) * 25 + tap] + f1 * lw[(0 * 64 + c + 1) * 25 + tap];
          a1 += f0 * lw[(1 * 64 + c) * 25 + tap] + f1 * lw[(1 * 64 + c + 1) * 25 + tap];
          a2 += f0 * lw[(2 * 64 + c) * 25 + tap] + f1 * lw[(2 * 64 + c + 1) * 25 + tap];
        }
      }
    }
  }
  out[((size_t)(b * 3 + 0) * H_LAT + h) * W_LON + w] = a0 + b3[0];
  out[((size_t)(b * 3 + 1) * H_LAT + h) * W_LON + w] = a1 + b3[1];
  out[((size_t)(b * 3 + 2) * H_LAT + h) * W_LON + w] = a2 + b3[2];
}

extern "C" void kernel_launch(void* const* d_in, const int* in_sizes, int n_in,
                              void* d_out, int out_size, void* d_ws, size_t ws_size,
                              hipStream_t stream) {
  const float* x    = (const float*)d_in[0];
  const float* psi  = (const float*)d_in[1];
  const float* quad = (const float*)d_in[2];
  const float* w1   = (const float*)d_in[3];
  const float* b1   = (const float*)d_in[4];
  const float* w2   = (const float*)d_in[5];
  const float* b2   = (const float*)d_in[6];
  const float* w3   = (const float*)d_in[7];
  const float* b3   = (const float*)d_in[8];
  (void)in_sizes; (void)n_in; (void)out_size; (void)ws_size;

  char* ws = (char*)d_ws;
  size_t off = 0;
  __bf16* z   = (__bf16*)(ws + off); off += (size_t)NSPAT * 32 * 2;   // 33.4 MB
  __bf16* h1  = (__bf16*)(ws + off); off += (size_t)NSPAT * 128 * 2;  // 133.4 MB
  __bf16* h2  = (__bf16*)(ws + off); off += (size_t)NSPAT * 64 * 2;   // 66.7 MB
  __bf16* w1p = (__bf16*)(ws + off); off += 128 * 32 * 2;
  __bf16* w2t = (__bf16*)(ws + off); off += (size_t)64 * 25 * 128 * 2;

  k_prep_w1<<<1, 128, 0, stream>>>(w1, w1p);
  k_prep_w2<<<800, 256, 0, stream>>>(w2, w2t);
  k_disco_z<<<(NSPAT * 32) / 256, 256, 0, stream>>>(x, psi, quad, z);
  k_gemm1<<<NSPAT / 64, 256, 0, stream>>>(z, w1p, b1, h1);

  dim3 g3(3, H_LAT, BATCH);
  const size_t lds3 = (size_t)(5 * 132 * 128 + 2 * 64 * 128) * sizeof(__bf16);  // 201,728 B
  k_conv2<<<g3, 256, lds3, stream>>>(h1, w2t, b2, h2);

  k_conv3<<<(NSPAT + 255) / 256, 256, 0, stream>>>(h2, w3, b3, (float*)d_out);
}
